// DotAttention_18494129177216
// MI455X (gfx1250) — matrix operand order repaired and checked by reading, not compiled, not run on hardware
//
#include <hip/hip_runtime.h>
#include <hip/hip_bf16.h>
#include <cmath>

// ---------------------------------------------------------------------------
// DotAttention for MI455X (gfx1250).
// All GEMMs via v_wmma_f32_16x16x32_f16, f32 accumulate. Operands are
// pre-converted to f16 so GEMM loops are pure global_load_b128 -> v_wmma
// (no LDS staging, no barriers, no conversions in the hot path).
// ---------------------------------------------------------------------------

typedef __attribute__((ext_vector_type(16))) _Float16 v16h;
typedef __attribute__((ext_vector_type(8)))  _Float16 v8h;
typedef __attribute__((ext_vector_type(8)))  float    v8f;

constexpr int Hdim = 1024;   // hidden
constexpr int Tt   = 512;    // target len
constexpr int Ss   = 512;    // source len
constexpr int NB   = 32;     // batch
constexpr int K2H  = 2 * Hdim;

// Build a 16-half WMMA operand from two contiguous 8-half (16B) chunks.
__device__ __forceinline__ v16h load_pair(const _Float16* p0, const _Float16* p1) {
  v8h lo = *(const v8h*)p0;
  v8h hi = *(const v8h*)p1;
  return __builtin_shufflevector(lo, hi, 0,1,2,3,4,5,6,7,8,9,10,11,12,13,14,15);
}

__device__ __forceinline__ v8f wmma_f16(v16h a, v16h b, v8f c) {
  return __builtin_amdgcn_wmma_f32_16x16x32_f16(false, a, false, b, (short)0, c,
                                                false, false);
}

// ---------------------------------------------------------------------------
// Kernel 0a: q (T,B,H) f32 -> f16 into the LOW half of xh[b][t][0:2H].
// grid (T, B), block 256.
// ---------------------------------------------------------------------------
__global__ void cvt_q_kernel(const float* __restrict__ q, _Float16* __restrict__ xh) {
  const int t = blockIdx.x, b = blockIdx.y;
  const float* src = q + ((size_t)t * NB + b) * Hdim;
  _Float16* dst = xh + ((size_t)b * Tt + t) * K2H;
#pragma unroll
  for (int rep = 0; rep < 4; ++rep) {
    const int i = threadIdx.x + rep * 256;
    dst[i] = (_Float16)src[i];
  }
}

// ---------------------------------------------------------------------------
// Kernel 0b: annotations (S,B,H) f32 -> ah[b][s][h] f16 and aTh[b][h][s] f16.
// grid (S/32, H/32, B), block 256.
// ---------------------------------------------------------------------------
__global__ void cvt_a_kernel(const float* __restrict__ a,
                             _Float16* __restrict__ ah,
                             _Float16* __restrict__ aTh) {
  __shared__ __align__(16) _Float16 tile[32][33];
  const int s0 = blockIdx.x * 32;
  const int h0 = blockIdx.y * 32;
  const int b  = blockIdx.z;
  const int c  = threadIdx.x & 31;
  const int r0 = threadIdx.x >> 5;
#pragma unroll
  for (int rr = 0; rr < 4; ++rr) {
    const int r = r0 + rr * 8;
    _Float16 hv = (_Float16)a[((size_t)(s0 + r) * NB + b) * Hdim + (h0 + c)];
    ah[((size_t)b * Ss + (s0 + r)) * Hdim + (h0 + c)] = hv;
    tile[r][c] = hv;
  }
  __syncthreads();
#pragma unroll
  for (int rr = 0; rr < 4; ++rr) {
    const int r = r0 + rr * 8;
    aTh[((size_t)b * Hdim + (h0 + r)) * Ss + (s0 + c)] = tile[c][r];
  }
}

// ---------------------------------------------------------------------------
// Kernel 0c: W (H x 2H) f32 -> f16.
// ---------------------------------------------------------------------------
__global__ void cvt_w_kernel(const float* __restrict__ W, _Float16* __restrict__ Wh) {
  size_t base = (size_t)blockIdx.x * 1024 + threadIdx.x;
#pragma unroll
  for (int rep = 0; rep < 4; ++rep)
    Wh[base + rep * 256] = (_Float16)W[base + rep * 256];
}

// ---------------------------------------------------------------------------
// Kernel 1: scores = softmax(mask(q @ a^T)).
// Block = 16 t-rows x full S=512 (8 waves x 64 s-cols, 4 accs each).
// K = H, step 64. All 8 B tiles preloaded before the WMMA chain so the
// scheduler can clause the loads and batch the loadcnt waits.
// grid (T/16, B), block 256.
// ---------------------------------------------------------------------------
__global__ void attn_scores_softmax(const _Float16* __restrict__ qh,  // low half of xh (B,T,2H)
                                    const _Float16* __restrict__ ah,  // (B,S,H)
                                    const int* __restrict__ lens,     // (B)
                                    float* __restrict__ scores_out,   // (T,B,S)
                                    _Float16* __restrict__ Ph) {      // (B,T,S)
  const int t0  = blockIdx.x * 16;
  const int b   = blockIdx.y;
  const int tid = threadIdx.x;
  const int lane = tid & 31;
  const int wv   = tid >> 5;
  const int len  = lens[b];

  __shared__ __align__(16) float sc[16 * 512];
  __shared__ __align__(16) float red[16 * 16];

  v8f acc[4] = {};
  const int m     = lane & 15;
  const int akoff = (lane < 16) ? 0 : 8;    // A: lanes 0-15 K 0-7/16-23, 16-31 K 8-15/24-31
  const int n     = lane & 15;
  const int bkoff = (lane < 16) ? 0 : 8;    // B: lanes 0-15 K 0-7/16-23, 16-31 K 8-15/24-31

  const _Float16* arow = qh + ((size_t)b * Tt + (t0 + m)) * K2H;  // q in cols [0,H)
  const _Float16* brow[4];
#pragma unroll
  for (int tt = 0; tt < 4; ++tt)
    brow[tt] = ah + ((size_t)b * Ss + (wv * 64 + tt * 16 + n)) * Hdim + bkoff;

  for (int k0 = 0; k0 < Hdim; k0 += 64) {
    __builtin_prefetch(arow + k0 + 256, 0, 3);
    v16h A0 = load_pair(arow + k0 + akoff,      arow + k0 + akoff + 16);
    v16h A1 = load_pair(arow + k0 + 32 + akoff, arow + k0 + 32 + akoff + 16);
    v16h Bv[4][2];
#pragma unroll
    for (int tt = 0; tt < 4; ++tt) {
      const _Float16* bp = brow[tt] + k0;
      Bv[tt][0] = load_pair(bp,      bp + 16);
      Bv[tt][1] = load_pair(bp + 32, bp + 48);
    }
#pragma unroll
    for (int tt = 0; tt < 4; ++tt) {
      acc[tt] = wmma_f16(A0, Bv[tt][0], acc[tt]);
      acc[tt] = wmma_f16(A1, Bv[tt][1], acc[tt]);
    }
  }

  // Spill accumulators: acc[tt] vgpr r, lane -> (row = r + 8*(lane>=16), col)
  {
    const int mbase = (lane < 16) ? 0 : 8;
#pragma unroll
    for (int tt = 0; tt < 4; ++tt) {
      const int col = wv * 64 + tt * 16 + (lane & 15);
#pragma unroll
      for (int r = 0; r < 8; ++r)
        sc[(mbase + r) * 512 + col] = acc[tt][r];
    }
  }
  __syncthreads();

  // Masked softmax over S per row: 16 threads/row, 32 cols each.
  const int row = tid >> 4;
  const int cg  = tid & 15;
  float mx = -INFINITY;
  for (int c = cg * 32; c < cg * 32 + 32; ++c) {
    float v = (c < len) ? sc[row * 512 + c] : -INFINITY;
    mx = fmaxf(mx, v);
  }
  red[row * 16 + cg] = mx;
  __syncthreads();
  if (cg == 0) {
    float m2 = -INFINITY;
    for (int i = 0; i < 16; ++i) m2 = fmaxf(m2, red[row * 16 + i]);
    red[row * 16] = m2;
  }
  __syncthreads();
  const float rmax = red[row * 16];
  __syncthreads();
  float ssum = 0.f;
  for (int c = cg * 32; c < cg * 32 + 32; ++c)
    if (c < len) ssum += __expf(sc[row * 512 + c] - rmax);
  red[row * 16 + cg] = ssum;
  __syncthreads();
  if (cg == 0) {
    float s2 = 0.f;
    for (int i = 0; i < 16; ++i) s2 += red[row * 16 + i];
    red[row * 16] = s2;
  }
  __syncthreads();
  const float inv = 1.0f / red[row * 16];

  float*    orow = scores_out + ((size_t)(t0 + row) * NB + b) * Ss;
  _Float16* prow = Ph + ((size_t)b * Tt + (t0 + row)) * Ss;
  for (int c = cg * 32; c < cg * 32 + 32; ++c) {
    float p = (c < len) ? __expf(sc[row * 512 + c] - rmax) * inv : 0.0f;
    orow[c] = p;
    prow[c] = (_Float16)p;
  }
}

// ---------------------------------------------------------------------------
// Kernel 2: contexts = P @ a, written into the HIGH half of xh[b][t][H:2H].
// Each wave computes a 32x64 tile (2 M-tiles x 4 N-tiles, 8 accumulators):
// every B pair feeds 4 WMMAs. No LDS, no barriers. K = S, step 64.
// grid (T/32, H/512, B), block 256.
// ---------------------------------------------------------------------------
__global__ void attn_context(const _Float16* __restrict__ Ph,   // (B,T,S)
                             const _Float16* __restrict__ aTh,  // (B,H,S)
                             _Float16* __restrict__ xh) {       // (B,T,2H)
  const int t0  = blockIdx.x * 32;
  const int h0  = blockIdx.y * 512;
  const int b   = blockIdx.z;
  const int lane = threadIdx.x & 31;
  const int wv   = threadIdx.x >> 5;

  v8f acc[2][4] = {};
  const int m     = lane & 15;
  const int akoff = (lane < 16) ? 0 : 8;
  const int n     = lane & 15;
  const int bkoff = (lane < 16) ? 0 : 8;

  const _Float16* arow0 = Ph + ((size_t)b * Tt + (t0 + m)) * Ss;
  const _Float16* arow1 = Ph + ((size_t)b * Tt + (t0 + 16 + m)) * Ss;

  for (int s0 = 0; s0 < Ss; s0 += 64) {
    __builtin_prefetch(arow0 + s0 + 256, 0, 3);
    v16h A00 = load_pair(arow0 + s0 + akoff,      arow0 + s0 + akoff + 16);
    v16h A01 = load_pair(arow0 + s0 + 32 + akoff, arow0 + s0 + 32 + akoff + 16);
    v16h A10 = load_pair(arow1 + s0 + akoff,      arow1 + s0 + akoff + 16);
    v16h A11 = load_pair(arow1 + s0 + 32 + akoff, arow1 + s0 + 32 + akoff + 16);
#pragma unroll
    for (int tt = 0; tt < 4; ++tt) {
      // B[k][n] = a[s0+k][h] = aTh[h][s0+k]: contiguous along s.
      const int h = h0 + wv * 64 + tt * 16 + n;
      const _Float16* bp = aTh + ((size_t)b * Hdim + h) * Ss + s0 + bkoff;
      v16h B0 = load_pair(bp,      bp + 16);
      v16h B1 = load_pair(bp + 32, bp + 48);
      acc[0][tt] = wmma_f16(A00, B0, acc[0][tt]);
      acc[1][tt] = wmma_f16(A10, B0, acc[1][tt]);
      acc[0][tt] = wmma_f16(A01, B1, acc[0][tt]);
      acc[1][tt] = wmma_f16(A11, B1, acc[1][tt]);
    }
  }

  const int mbase = (lane < 16) ? 0 : 8;
#pragma unroll
  for (int mt = 0; mt < 2; ++mt) {
#pragma unroll
    for (int tt = 0; tt < 4; ++tt) {
      const int h = h0 + wv * 64 + tt * 16 + (lane & 15);
#pragma unroll
      for (int r = 0; r < 8; ++r)
        xh[((size_t)b * Tt + (t0 + mt * 16 + mbase + r)) * K2H + Hdim + h] =
            (_Float16)acc[mt][tt][r];
    }
  }
}

// ---------------------------------------------------------------------------
// Kernel 3: attentional = tanh(X @ W^T + bias), X = xh (concat pre-fused).
// Same 32x64 per-wave tiling. No LDS, no barriers. K = 2H, step 64.
// grid (T/32, H/512, B), block 256.
// ---------------------------------------------------------------------------
__global__ void attn_out_kernel(const _Float16* __restrict__ xh,   // (B,T,2H)
                                const _Float16* __restrict__ Wh,   // (H,2H)
                                const float* __restrict__ bias,    // (H)
                                float* __restrict__ out) {         // (T,B,H)
  const int t0  = blockIdx.x * 32;
  const int h0  = blockIdx.y * 512;
  const int b   = blockIdx.z;
  const int lane = threadIdx.x & 31;
  const int wv   = threadIdx.x >> 5;

  v8f acc[2][4] = {};
  const int m     = lane & 15;
  const int akoff = (lane < 16) ? 0 : 8;
  const int n     = lane & 15;
  const int bkoff = (lane < 16) ? 0 : 8;

  const _Float16* arow0 = xh + ((size_t)b * Tt + (t0 + m)) * K2H;
  const _Float16* arow1 = xh + ((size_t)b * Tt + (t0 + 16 + m)) * K2H;

  for (int k0 = 0; k0 < K2H; k0 += 64) {
    __builtin_prefetch(arow0 + k0 + 256, 0, 3);
    v16h A00 = load_pair(arow0 + k0 + akoff,      arow0 + k0 + akoff + 16);
    v16h A01 = load_pair(arow0 + k0 + 32 + akoff, arow0 + k0 + 32 + akoff + 16);
    v16h A10 = load_pair(arow1 + k0 + akoff,      arow1 + k0 + akoff + 16);
    v16h A11 = load_pair(arow1 + k0 + 32 + akoff, arow1 + k0 + 32 + akoff + 16);
#pragma unroll
    for (int tt = 0; tt < 4; ++tt) {
      // B[k][n] = W[h_out][k0+k]: contiguous along W's row.
      const int hout = h0 + wv * 64 + tt * 16 + n;
      const _Float16* bp = Wh + (size_t)hout * K2H + k0 + bkoff;
      v16h B0 = load_pair(bp,      bp + 16);
      v16h B1 = load_pair(bp + 32, bp + 48);
      acc[0][tt] = wmma_f16(A00, B0, acc[0][tt]);
      acc[1][tt] = wmma_f16(A10, B0, acc[1][tt]);
      acc[0][tt] = wmma_f16(A01, B1, acc[0][tt]);
      acc[1][tt] = wmma_f16(A11, B1, acc[1][tt]);
    }
  }

  const int mbase = (lane < 16) ? 0 : 8;
#pragma unroll
  for (int mt = 0; mt < 2; ++mt) {
#pragma unroll
    for (int tt = 0; tt < 4; ++tt) {
      const int h = h0 + wv * 64 + tt * 16 + (lane & 15);
      const float bb = bias[h];
#pragma unroll
      for (int r = 0; r < 8; ++r)
        out[((size_t)(t0 + mt * 16 + mbase + r) * NB + b) * Hdim + h] =
            tanhf(acc[mt][tt][r] + bb);
    }
  }
}

// ---------------------------------------------------------------------------
extern "C" void kernel_launch(void* const* d_in, const int* in_sizes, int n_in,
                              void* d_out, int out_size, void* d_ws, size_t ws_size,
                              hipStream_t stream) {
  (void)in_sizes; (void)n_in; (void)out_size; (void)ws_size;

  const float* q    = (const float*)d_in[0];  // (T,B,H)
  const float* a    = (const float*)d_in[1];  // (S,B,H)
  const int*   lens = (const int*)d_in[2];    // (B)
  const float* W    = (const float*)d_in[3];  // (H,2H)
  const float* bias = (const float*)d_in[4];  // (H)

  float* out_attn   = (float*)d_out;                        // T*B*H
  float* out_scores = out_attn + (size_t)Tt * NB * Hdim;    // T*B*S

  // Workspace (f16): ah 32MB, aTh 32MB, Wh 4MB, xh 64MB, Ph 16MB = 148MB
  _Float16* ah  = (_Float16*)d_ws;                          // (B,S,H)
  _Float16* aTh = ah  + (size_t)NB * Ss * Hdim;             // (B,H,S)
  _Float16* Wh  = aTh + (size_t)NB * Hdim * Ss;             // (H,2H)
  _Float16* xh  = Wh  + (size_t)Hdim * K2H;                 // (B,T,2H) = [q | ctx]
  _Float16* Ph  = xh  + (size_t)NB * Tt * K2H;              // (B,T,S)

  cvt_q_kernel<<<dim3(Tt, NB), 256, 0, stream>>>(q, xh);
  cvt_a_kernel<<<dim3(Ss / 32, Hdim / 32, NB), 256, 0, stream>>>(a, ah, aTh);
  cvt_w_kernel<<<dim3((Hdim * K2H) / 1024), 256, 0, stream>>>(W, Wh);
  attn_scores_softmax<<<dim3(Tt / 16, NB), 256, 0, stream>>>(xh, ah, lens, out_scores, Ph);
  attn_context<<<dim3(Tt / 32, Hdim / 512, NB), 256, 0, stream>>>(Ph, aTh, xh);
  attn_out_kernel<<<dim3(Tt / 32, Hdim / 512, NB), 256, 0, stream>>>(xh, Wh, bias, out_attn);
}
